// DTW_56272661512310
// MI455X (gfx1250) — matrix operand order repaired and checked by reading, not compiled, not run on hardware
//
#include <hip/hip_runtime.h>
#include <hip/hip_bf16.h>
#include <math.h>

// Problem constants (match reference)
#define BB 64
#define II 512
#define JJ 384
#define DD 768

typedef __attribute__((ext_vector_type(16))) _Float16 v16h;
typedef __attribute__((ext_vector_type(8)))  _Float16 v8h;
typedef __attribute__((ext_vector_type(8)))  float    v8f;
typedef __attribute__((ext_vector_type(4)))  unsigned v4u;
typedef __attribute__((ext_vector_type(8)))  unsigned v8u;

// ---------------------------------------------------------------------------
// CDNA5 Tensor Data Mover: one instruction DMAs a whole 2D tile to LDS.
// D# built per CDNA5 ISA (2-operand form, 2D tensor).
// ---------------------------------------------------------------------------
__device__ __forceinline__ v8u tdm_group1(unsigned tensor_d0, unsigned tensor_d1,
                                          unsigned tile_d0, unsigned tile_d1,
                                          unsigned stride0) {
    v8u g;
    g[0] = (1u << 16);                                   // mask=0, data_size=2B
    g[1] = (tensor_d0 & 0xFFFFu) << 16;                  // dim0[15:0] @ bits 63:48
    g[2] = (tensor_d0 >> 16) | ((tensor_d1 & 0xFFFFu) << 16);
    g[3] = (tensor_d1 >> 16) | (tile_d0 << 16);          // tile_dim0 @ 127:112
    g[4] = tile_d1;                                      // tile_dim1; tile_dim2=0
    g[5] = stride0;                                      // dim0_stride[31:0]
    g[6] = 0;                                            // stride hi / dim1_stride
    g[7] = 0;
    return g;
}

__device__ __forceinline__ void tdm_load(unsigned lds_addr,
                                         unsigned long long gaddr,
                                         v8u g1) {
    v4u g0;
    g0[0] = 1u;                                          // count=1 (valid), user D#
    g0[1] = lds_addr;                                    // LDS byte address
    g0[2] = (unsigned)gaddr;                             // global addr [31:0]
    g0[3] = ((unsigned)(gaddr >> 32) & 0x01FFFFFFu) | (2u << 30); // [56:32]|type=2
    asm volatile("tensor_load_to_lds %0, %1" :: "s"(g0), "s"(g1) : "memory");
}

__device__ __forceinline__ void wait_tensor0() {
#if __has_builtin(__builtin_amdgcn_s_wait_tensorcnt)
    __builtin_amdgcn_s_wait_tensorcnt(0);
#else
    asm volatile("s_wait_tensorcnt 0" ::: "memory");
#endif
}

// CDNA5 hardware tanh (V_TANH_F32)
__device__ __forceinline__ float fast_tanhf(float x) {
#if __has_builtin(__builtin_amdgcn_tanhf)
    return __builtin_amdgcn_tanhf(x);
#elif __has_builtin(__builtin_amdgcn_tanh_f32)
    return __builtin_amdgcn_tanh_f32(x);
#else
    float r;
    asm("v_tanh_f32 %0, %1\n\tv_nop" : "=v"(r) : "v"(x));
    return r;
#endif
}

// ---------------------------------------------------------------------------
// Kernel 1: L2-normalize rows of length 768 (f32 in) -> f16 out
// ---------------------------------------------------------------------------
__global__ void normalize_f16_kernel(const float* __restrict__ src,
                                     _Float16* __restrict__ dst) {
    const float*  row  = src + (size_t)blockIdx.x * DD;
    _Float16*     orow = dst + (size_t)blockIdx.x * DD;
    int t = threadIdx.x;

    float x0 = row[t];
    float x1 = row[t + 256];
    float x2 = row[t + 512];
    float s  = x0 * x0 + x1 * x1 + x2 * x2;

    #pragma unroll
    for (int off = 16; off > 0; off >>= 1)
        s += __shfl_down(s, off, 32);

    __shared__ float smem[8];
    int lane = t & 31, wid = t >> 5;
    if (lane == 0) smem[wid] = s;
    __syncthreads();
    if (wid == 0) {
        float v = (lane < 8) ? smem[lane] : 0.0f;
        #pragma unroll
        for (int off = 4; off > 0; off >>= 1)
            v += __shfl_down(v, off, 32);
        if (lane == 0) smem[0] = v;
    }
    __syncthreads();

    float inv = 1.0f / fmaxf(sqrtf(smem[0]), 1e-8f);
    orow[t]       = (_Float16)(x0 * inv);
    orow[t + 256] = (_Float16)(x1 * inv);
    orow[t + 512] = (_Float16)(x2 * inv);
}

// ---------------------------------------------------------------------------
// Kernel 2: batched GEMM  sim[b,i,j] = tanh(a * <n1[b,i,:], n2[b,j,:]> + b)
// Block = 256 threads (8 waves), macro-tile 128(i) x 64(j), K step 32.
// Double-buffered LDS staged by the Tensor Data Mover (wave 0 issues two
// tensor_load_to_lds per k-step: A panel 128x32 f16, B panel 64x32 f16, row
// stride 768), overlapping the 4 WMMAs per wave from the other buffer.
// TDM packs rows contiguously: row y -> LDS offset y*64 B (matches reads).
// ---------------------------------------------------------------------------
#define ABYTES 8192              // 128 rows * 64 B
#define BBYTES 4096              // 64 rows * 64 B
#define BUFBYTES (ABYTES + BBYTES)

__global__ void sim_gemm_kernel(const _Float16* __restrict__ n1h,
                                const _Float16* __restrict__ n2h,
                                float* __restrict__ sim,
                                const float* __restrict__ pa,
                                const float* __restrict__ pb) {
    extern __shared__ _Float16 smem_h[];                          // 2 * 12 KB
    const unsigned ldsBase = __builtin_amdgcn_groupstaticsize();  // dyn LDS start

    const int t    = threadIdx.x;
    const int w    = t >> 5;                   // wave id: i-tile 0..7
    const int lane = t & 31;
    const int mn   = lane & 15;
    const int g    = lane >> 4;

    const int jS = blockIdx.x % 6;             // j strip of 64
    const int iS = (blockIdx.x / 6) % 4;       // i strip of 128
    const int bb = blockIdx.x / 24;            // batch

    const unsigned long long aBase0 =
        (unsigned long long)(n1h + ((size_t)bb * II + (size_t)iS * 128) * DD);
    const unsigned long long bBase0 =
        (unsigned long long)(n2h + ((size_t)bb * JJ + (size_t)jS * 64) * DD);

    // tile-invariant D# group1 descriptors (uniform -> SGPRs)
    const v8u g1A = tdm_group1(/*tensor*/32, 128, /*tile*/32, 128, /*stride*/DD);
    const v8u g1B = tdm_group1(/*tensor*/32,  64, /*tile*/32,  64, /*stride*/DD);

    v8f acc[4];
    #pragma unroll
    for (int jj = 0; jj < 4; ++jj) acc[jj] = (v8f){0,0,0,0,0,0,0,0};

    union Frag { v16h v; v8h h[2]; };

    auto stage = [&](int kNext, int buf) {     // wave 0 only
        const unsigned aL = ldsBase + (unsigned)(buf * BUFBYTES);
        tdm_load(aL,          aBase0 + (unsigned long long)kNext * 2, g1A);
        tdm_load(aL + ABYTES, bBase0 + (unsigned long long)kNext * 2, g1B);
    };

    auto compute = [&](int buf) {
        const _Float16* aLdsP = smem_h + (size_t)buf * (BUFBYTES / 2);
        const _Float16* bLdsP = aLdsP + (ABYTES / 2);

        Frag A;
        const _Float16* Ar = aLdsP + (w * 16 + mn) * 32;
        A.h[0] = *(const v8h*)(Ar + 8 * g);
        A.h[1] = *(const v8h*)(Ar + 16 + 8 * g);
        Frag Bm[4];
        #pragma unroll
        for (int jj = 0; jj < 4; ++jj) {
            const _Float16* Br = bLdsP + (jj * 16 + mn) * 32;
            Bm[jj].h[0] = *(const v8h*)(Br + 8 * g);
            Bm[jj].h[1] = *(const v8h*)(Br + 16 + 8 * g);
        }
        #pragma unroll
        for (int jj = 0; jj < 4; ++jj)
            acc[jj] = __builtin_amdgcn_wmma_f32_16x16x32_f16(
                false, A.v, false, Bm[jj].v, (short)0, acc[jj], false, false);
    };

    // ---- software pipeline: TDM stage(k+32) overlaps compute(k) -----------
    if (w == 0) { stage(0, 0); wait_tensor0(); }
    __syncthreads();

    for (int k = 0; k < DD; k += 64) {
        if (w == 0) stage(k + 32, 1);          // k+32 <= 736 < 768 always
        compute(0);                            // consume buf0 @ k
        if (w == 0) wait_tensor0();
        __syncthreads();

        if (w == 0 && k + 64 < DD) stage(k + 64, 0);
        compute(1);                            // consume buf1 @ k+32
        if (w == 0) wait_tensor0();
        __syncthreads();
    }

    const float aS = *pa;
    const float bS = *pb;

    #pragma unroll
    for (int jj = 0; jj < 4; ++jj) {
        const int jcol = jS * 64 + jj * 16 + mn;
        #pragma unroll
        for (int r = 0; r < 8; ++r) {
            const int irow = iS * 128 + w * 16 + r + 8 * g;
            sim[((size_t)bb * II + irow) * JJ + jcol] =
                fast_tanhf(aS * acc[jj][r] + bS);
        }
    }
}

// ---------------------------------------------------------------------------
// Kernel 3: DTW-style column scan (one 512-thread block per batch)
// ---------------------------------------------------------------------------
__global__ void dtw_kernel(const float* __restrict__ sim,
                           float* __restrict__ out) {
    const int b    = blockIdx.x;
    const int i    = threadIdx.x;
    const int lane = i & 31;
    const int wid  = i >> 5;                 // 16 waves

    const float* S = sim + (size_t)b * II * JJ;
    __shared__ float wmax[16];

    float v = S[(size_t)i * JJ];             // p0 = sim[:,0]

    for (int j = 1; j < JJ; ++j) {
        float c = v;
        #pragma unroll
        for (int off = 1; off < 32; off <<= 1) {
            float o = __shfl_up(c, off, 32);
            if (lane >= off) c = fmaxf(c, o);
        }
        if (lane == 31) wmax[wid] = c;
        __syncthreads();
        if (wid == 0) {
            float tt = (lane < 16) ? wmax[lane] : -1e30f;
            #pragma unroll
            for (int off = 1; off < 16; off <<= 1) {
                float o = __shfl_up(tt, off, 32);
                if (lane >= off) tt = fmaxf(tt, o);
            }
            if (lane < 16) wmax[lane] = tt;
        }
        __syncthreads();

        float base = (wid > 0) ? wmax[wid - 1] : -1e30f;
        float incl = fmaxf(c, base);
        float excl = __shfl_up(incl, 1, 32);
        if (lane == 0) excl = base;

        float r = fmaxf(S[(size_t)i * JJ + j], 0.0f);
        v = (i == 0) ? r : (excl + r);
        __syncthreads();
    }

    if (i == II - 1) out[b] = v / (float)JJ;
}

// ---------------------------------------------------------------------------
// Host launcher
// ---------------------------------------------------------------------------
extern "C" void kernel_launch(void* const* d_in, const int* in_sizes, int n_in,
                              void* d_out, int out_size, void* d_ws, size_t ws_size,
                              hipStream_t stream) {
    const float* emb1 = (const float*)d_in[0];   // (B,I,D) f32
    const float* emb2 = (const float*)d_in[1];   // (B,J,D) f32
    const float* pa   = (const float*)d_in[2];   // scalar
    const float* pb   = (const float*)d_in[3];   // scalar
    float* out = (float*)d_out;                  // (B,) f32

    // Workspace: n1h (f16) | n2h (f16) | sim (f32)   ~138 MB
    _Float16* n1h = (_Float16*)d_ws;
    _Float16* n2h = n1h + (size_t)BB * II * DD;
    float*    sim = (float*)(n2h + (size_t)BB * JJ * DD);

    normalize_f16_kernel<<<BB * II, 256, 0, stream>>>(emb1, n1h);
    normalize_f16_kernel<<<BB * JJ, 256, 0, stream>>>(emb2, n2h);

    // 64 batches * 4 i-strips(128) * 6 j-strips(64) = 1536 blocks, 24 KB LDS
    sim_gemm_kernel<<<BB * 4 * 6, 256, 2 * BUFBYTES, stream>>>(n1h, n2h, sim, pa, pb);

    dtw_kernel<<<BB, 512, 0, stream>>>(sim, out);
}
